// LstmForClassification_45586782880213
// MI455X (gfx1250) — compile-verified, matrix operand order checked by reading, not run on hardware
//
#include <hip/hip_runtime.h>
#include <stdint.h>

#define HIDDEN   512
#define NUMLBL   4
#define BB       32
#define TT       2048

typedef __attribute__((ext_vector_type(16))) __bf16        bfrag16;
typedef __attribute__((ext_vector_type(8)))  float         f32x8;
typedef __attribute__((ext_vector_type(8)))  unsigned short us8_t;

union FragU { us8_t h[2]; bfrag16 v; };

// ---------- helpers ----------
__device__ __forceinline__ unsigned short f2bf(float f) {
    unsigned u = __float_as_uint(f);
    u += 0x7FFFu + ((u >> 16) & 1u);          // round-to-nearest-even
    return (unsigned short)(u >> 16);
}
__device__ __forceinline__ float bf2f(unsigned short h) {
    return __uint_as_float(((unsigned)h) << 16);
}
__device__ __forceinline__ float sigm(float x) {
    return 1.0f / (1.0f + __expf(-x));
}

// 16-bit A/B fragment for v_wmma_f32_16x16x32_bf16, sourced from an LDS
// row-major [rows][512] bf16 tile. Lane L<16: row=row0+L, K={k0..k0+7, k0+16..k0+23};
// lane L>=16: row=row0+L-16, K={k0+8..k0+15, k0+24..k0+31}.  => 2x ds_load_b128.
__device__ __forceinline__ bfrag16 load_frag(const unsigned short* base,
                                             int row0, int k0, int lane) {
    int r  = row0 + (lane & 15);
    int kb = k0 + ((lane & 16) ? 8 : 0);
    const unsigned short* p = base + r * HIDDEN + kb;
    FragU u;
    u.h[0] = *(const us8_t*)(p);
    u.h[1] = *(const us8_t*)(p + 16);
    return u.v;
}

// ---------- kernel 0: zero barrier counters ----------
__global__ void init_kernel(unsigned int* cnt) {
    if (threadIdx.x < 2) cnt[threadIdx.x] = 0u;
}

// ---------- kernel 1: embedding gather, fp32 -> bf16, time-major ----------
// grid = T*B blocks (n = t*B + b), 256 threads, 2 elems/thread
__global__ void embed_kernel(const int* __restrict__ x,
                             const float* __restrict__ emb,
                             unsigned short* __restrict__ X0) {
    int n = blockIdx.x;
    int t = n >> 5;            // /B
    int b = n & 31;            // %B
    int tok = x[b * TT + t];
    const float2* src = (const float2*)(emb + (size_t)tok * HIDDEN);
    unsigned* dst = (unsigned*)(X0 + (size_t)n * HIDDEN);
    float2 v = src[threadIdx.x];
    dst[threadIdx.x] = (unsigned)f2bf(v.x) | ((unsigned)f2bf(v.y) << 16);
}

// ---------- kernel 2: persistent fused LSTM layer ----------
// 32 WGs x 256 threads; WG w owns hidden units [w*16, w*16+16) -> 64 gate rows.
// Both weight slices resident in LDS as bf16; per step two [32x64]x[512] GEMMs
// on v_wmma_f32_16x16x32_bf16 with software-pipelined LDS fragment loads,
// fp32 gate math, bf16 h exchange via L2-resident double-buffered global h
// + software grid barrier.
#define SM_WIH   0
#define SM_WHH   65536
#define SM_HS    131072
#define SM_XS    163840     // aliased by gbuf after GEMMs
#define SM_CST   196608
#define SM_BIAS  198656
#define SM_TOTAL 198912

__global__ void __launch_bounds__(256, 1)
lstm_layer_kernel(const unsigned short* __restrict__ Xin,
                  unsigned short* __restrict__ Xout,
                  const float* __restrict__ Wih,
                  const float* __restrict__ Whh,
                  const float* __restrict__ bih,
                  const float* __restrict__ bhh,
                  unsigned short* __restrict__ hbuf,   // [2][B*H] bf16
                  unsigned int* __restrict__ cnt) {
    extern __shared__ char smem[];
    unsigned short* wih_s  = (unsigned short*)(smem + SM_WIH);   // [64][512]
    unsigned short* whh_s  = (unsigned short*)(smem + SM_WHH);   // [64][512]
    unsigned short* hs     = (unsigned short*)(smem + SM_HS);    // [32][512]
    unsigned short* xs     = (unsigned short*)(smem + SM_XS);    // [32][512]
    float*          gbuf   = (float*)(smem + SM_XS);             // [4][32][16] alias
    float*          cst    = (float*)(smem + SM_CST);            // [32][16]
    float*          bias_s = (float*)(smem + SM_BIAS);           // [64]

    const int tid = threadIdx.x;
    const int wg  = blockIdx.x;          // 0..31
    const int j0  = wg * 16;             // hidden-unit base

    // preload weight slices (f32 -> bf16) and bias, zero state
    for (int idx = tid; idx < 64 * HIDDEN; idx += 256) {
        int r = idx >> 9, k = idx & 511;
        int gr = (r >> 4) * HIDDEN + j0 + (r & 15);   // gate row: gt*512 + j0 + jj
        wih_s[idx] = f2bf(Wih[(size_t)gr * HIDDEN + k]);
        whh_s[idx] = f2bf(Whh[(size_t)gr * HIDDEN + k]);
    }
    for (int idx = tid; idx < 64; idx += 256) {
        int gr = (idx >> 4) * HIDDEN + j0 + (idx & 15);
        bias_s[idx] = bih[gr] + bhh[gr];
    }
    for (int idx = tid; idx < BB * 16; idx += 256) cst[idx] = 0.0f;
    for (int idx = tid; idx < BB * HIDDEN; idx += 256) hs[idx] = 0;
    __syncthreads();

    const int lane = tid & 31;
    const int wv   = tid >> 5;           // 0..7 (8 waves)
    const int m0   = (wv & 1) << 4;      // batch tile base
    const int gt   = wv >> 1;            // gate type 0..3 (i,f,g,o)
    const int row0 = gt << 4;            // row base inside 64-row slice
    const float bias = bias_s[row0 + (lane & 15)];

    for (int t = 0; t < TT; ++t) {
        // stage x_t (full [32][512] bf16) into LDS
        const us8_t* xsrc = (const us8_t*)(Xin + (size_t)t * BB * HIDDEN);
        us8_t* xdst = (us8_t*)xs;
        for (int v = tid; v < (BB * HIDDEN) / 8; v += 256) xdst[v] = xsrc[v];
        __syncthreads();

        // ---- software-pipelined GEMMs: loads for step k+1 issue before wmma(k)
        f32x8 acc = { bias, bias, bias, bias, bias, bias, bias, bias };
        bfrag16 a_cur = load_frag(xs,    m0,   0, lane);
        bfrag16 b_cur = load_frag(wih_s, row0, 0, lane);
        #pragma unroll
        for (int k0 = 0; k0 < HIDDEN; k0 += 32) {
            bfrag16 a_nxt, b_nxt;
            if (k0 + 32 < HIDDEN) {
                a_nxt = load_frag(xs,    m0,   k0 + 32, lane);
                b_nxt = load_frag(wih_s, row0, k0 + 32, lane);
            } else {            // bridge prefetch into the h @ Whh^T GEMM
                a_nxt = load_frag(hs,    m0,   0, lane);
                b_nxt = load_frag(whh_s, row0, 0, lane);
            }
            acc = __builtin_amdgcn_wmma_f32_16x16x32_bf16(
                      false, a_cur, false, b_cur, (short)0, acc, false, false);
            a_cur = a_nxt; b_cur = b_nxt;
        }
        #pragma unroll
        for (int k0 = 0; k0 < HIDDEN; k0 += 32) {
            bfrag16 a_nxt = a_cur, b_nxt = b_cur;
            if (k0 + 32 < HIDDEN) {
                a_nxt = load_frag(hs,    m0,   k0 + 32, lane);
                b_nxt = load_frag(whh_s, row0, k0 + 32, lane);
            }
            acc = __builtin_amdgcn_wmma_f32_16x16x32_bf16(
                      false, a_cur, false, b_cur, (short)0, acc, false, false);
            a_cur = a_nxt; b_cur = b_nxt;
        }
        __syncthreads();   // all waves done reading xs -> safe to alias gbuf

        {   // D layout: vgpr r, lane L -> M = m0 + r + (L>=16 ? 8:0), N = L&15
            int moff = m0 + ((lane & 16) ? 8 : 0);
            int jj = lane & 15;
            #pragma unroll
            for (int r = 0; r < 8; ++r)
                gbuf[(gt * 32 + moff + r) * 16 + jj] = acc[r];
        }
        __syncthreads();

        // fp32 gate math + state update; emit bf16 h slice
        for (int idx = tid; idx < BB * 16; idx += 256) {
            int b  = idx >> 4, jj = idx & 15;
            float gi = gbuf[(0 * 32 + b) * 16 + jj];
            float gf = gbuf[(1 * 32 + b) * 16 + jj];
            float gg = gbuf[(2 * 32 + b) * 16 + jj];
            float go = gbuf[(3 * 32 + b) * 16 + jj];
            float c  = sigm(gf) * cst[idx] + sigm(gi) * tanhf(gg);
            float h  = sigm(go) * tanhf(c);
            cst[idx] = c;
            unsigned short hb = f2bf(h);
            int col = j0 + jj;
            hbuf[(size_t)(t & 1) * (BB * HIDDEN) + b * HIDDEN + col] = hb;
            Xout[((size_t)t * BB + b) * HIDDEN + col] = hb;
        }

        // prefetch next x tile while we wait at the grid barrier
        if (t + 1 < TT) {
            const char* pf = (const char*)(Xin + (size_t)(t + 1) * BB * HIDDEN);
            __builtin_prefetch(pf + tid * 128, 0, 0);   // global_prefetch_b8
        }

        // software grid barrier across the 32 persistent WGs
        __threadfence();
        __syncthreads();
        if (tid == 0) {
            atomicAdd(cnt, 1u);
            unsigned target = (unsigned)(t + 1) * 32u;
            while (*(volatile unsigned*)cnt < target) __builtin_amdgcn_s_sleep(1);
            __threadfence();
        }
        __syncthreads();

        // stage h_t for next step (synced by next iteration's barrier)
        const us8_t* hsrc = (const us8_t*)(hbuf + (size_t)(t & 1) * BB * HIDDEN);
        us8_t* hdst = (us8_t*)hs;
        for (int v = tid; v < (BB * HIDDEN) / 8; v += 256) hdst[v] = hsrc[v];
    }
}

// ---------- kernel 3: classifier head + log-softmax + mean NLL ----------
__global__ void head_kernel(const unsigned short* __restrict__ hT,  // [B][H] bf16
                            const float* __restrict__ fc_w,         // [4][H]
                            const float* __restrict__ fc_b,         // [4]
                            const int* __restrict__ labels,         // [B]
                            float* __restrict__ out) {              // [1 + B*4]
    __shared__ float logits_s[BB][NUMLBL];
    __shared__ float nll[BB];
    int tid = threadIdx.x;            // 128 threads
    int b = tid >> 2, l = tid & 3;
    float s = fc_b[l];
    for (int k = 0; k < HIDDEN; ++k)
        s += bf2f(hT[b * HIDDEN + k]) * fc_w[l * HIDDEN + k];
    logits_s[b][l] = s;
    out[1 + b * NUMLBL + l] = s;
    __syncthreads();
    if (l == 0) {
        float m = logits_s[b][0];
        for (int j = 1; j < NUMLBL; ++j) m = fmaxf(m, logits_s[b][j]);
        float z = 0.0f;
        for (int j = 0; j < NUMLBL; ++j) z += __expf(logits_s[b][j] - m);
        float lse = m + logf(z);
        nll[b] = lse - logits_s[b][labels[b]];
    }
    __syncthreads();
    if (tid == 0) {
        float a = 0.0f;
        for (int j = 0; j < BB; ++j) a += nll[j];
        out[0] = a / (float)BB;
    }
}

// ---------- launcher ----------
extern "C" void kernel_launch(void* const* d_in, const int* in_sizes, int n_in,
                              void* d_out, int out_size, void* d_ws, size_t ws_size,
                              hipStream_t stream) {
    const int*   x      = (const int*)d_in[0];
    const int*   labels = (const int*)d_in[1];
    const float* emb    = (const float*)d_in[2];
    const float* W_ih   = (const float*)d_in[3];   // [2][2048][512]
    const float* W_hh   = (const float*)d_in[4];
    const float* b_ih   = (const float*)d_in[5];   // [2][2048]
    const float* b_hh   = (const float*)d_in[6];
    const float* fc_w   = (const float*)d_in[7];
    const float* fc_b   = (const float*)d_in[8];
    float* out = (float*)d_out;

    const size_t XBYTES = (size_t)TT * BB * HIDDEN * sizeof(unsigned short); // 64 MB
    char* ws = (char*)d_ws;
    unsigned short* X0   = (unsigned short*)(ws);
    unsigned short* X1   = (unsigned short*)(ws + XBYTES);
    unsigned short* hbuf = (unsigned short*)(ws + 2 * XBYTES);               // 64 KB
    unsigned int*   cnt  = (unsigned int*)(ws + 2 * XBYTES + 2 * BB * HIDDEN * sizeof(unsigned short));

    init_kernel<<<1, 64, 0, stream>>>(cnt);
    embed_kernel<<<TT * BB, 256, 0, stream>>>(x, emb, X0);

    const size_t WL = (size_t)4 * HIDDEN * HIDDEN;  // per-layer weight elems
    // layer 0: X0 -> X1
    lstm_layer_kernel<<<32, 256, SM_TOTAL, stream>>>(
        X0, X1, W_ih, W_hh, b_ih, b_hh, hbuf, cnt + 0);
    // layer 1: X1 -> X0
    lstm_layer_kernel<<<32, 256, SM_TOTAL, stream>>>(
        X1, X0, W_ih + WL, W_hh + WL, b_ih + 4 * HIDDEN, b_hh + 4 * HIDDEN,
        hbuf, cnt + 1);

    head_kernel<<<1, 128, 0, stream>>>(
        X0 + (size_t)(TT - 1) * BB * HIDDEN, fc_w, fc_b, labels, out);
    (void)in_sizes; (void)n_in; (void)out_size; (void)ws_size;
}
